// StructGNN_Model_19138374271351
// MI455X (gfx1250) — matrix operand order repaired and checked by reading, not compiled, not run on hardware
//
#include <hip/hip_runtime.h>
#include <cstdint>
#include <cstddef>

#define DEV_INLINE __device__ __forceinline__

typedef __bf16 v16bf __attribute__((ext_vector_type(16)));
typedef __bf16 v8bf  __attribute__((ext_vector_type(8)));
typedef __bf16 v4bf  __attribute__((ext_vector_type(4)));
typedef float  v8f   __attribute__((ext_vector_type(8)));
typedef float  v4f   __attribute__((ext_vector_type(4)));

static constexpr int Bb   = 8;
static constexpr int Nn   = 1024;
static constexpr int Kk   = 30;
static constexpr int Hh   = 128;
static constexpr int VOC  = 21;
static constexpr int Ll   = 3;
static constexpr float EPSC  = 1e-5f;
static constexpr float SCALE = 30.0f;
static constexpr int BN = Bb * Nn;  // 8192

// ---------------------------------------------------------------------------
// WMMA fragment helpers (layouts per CDNA5 ISA 7.12.2, wave32)
// ---------------------------------------------------------------------------

// A fragment: 16x32 bf16 tile from LDS, row-major, leading dim ld.
// Per-lane data is two contiguous 16-byte runs -> 2x ds_load_b128.
DEV_INLINE v16bf load_a(const __bf16* A, int ld, int mBase, int kBase, int lane) {
  const int half = lane >> 4;
  const int m    = lane & 15;
  const __bf16* p = A + (size_t)(mBase + m) * ld + kBase + half * 8;
  v8bf lo = *(const v8bf*)p;         // k = kBase + half*8 + 0..7
  v8bf hi = *(const v8bf*)(p + 16);  // k = kBase + 16 + half*8 + 0..7
  v16bf r;
#pragma unroll
  for (int i = 0; i < 8; ++i) {
    r[i]     = lo[i];
    r[i + 8] = hi[i];
  }
  return r;
}

// B fragment from pre-swizzled (packed) weights: lane's 16 values contiguous.
// Packed index: (((kt*numNT + nt)*32 + lane)*16 + e), e -> k_local = e<8 ? half*8+e : 16+half*8+(e-8).
DEV_INLINE v16bf load_bp(const __bf16* P, int numNT, int kt, int nt, int lane) {
  const __bf16* p = P + (((size_t)(kt * numNT + nt)) * 32 + lane) * 16;
  return *(const v16bf*)p;  // 32B aligned -> 2x global_load_b128
}

DEV_INLINE v8f wmma_bf16(v16bf a, v16bf b, v8f c) {
  return __builtin_amdgcn_wmma_f32_16x16x32_bf16(false, a, false, b, (short)0, c,
                                                 false, false);
}

// Store 16x16 f32 accumulator tile as bf16 with bias + ReLU into LDS.
DEV_INLINE void store_tile_relu(__bf16* C, int ld, int mBase, int nBase, v8f acc,
                                const float* bias, int lane) {
  const int half = lane >> 4;
  const int n    = lane & 15;
  const float bn = bias[nBase + n];
#pragma unroll
  for (int i = 0; i < 8; ++i) {
    float x = fmaxf(acc[i] + bn, 0.0f);
    C[(size_t)(mBase + half * 8 + i) * ld + nBase + n] = (__bf16)x;
  }
}

// ---------------------------------------------------------------------------
// Weight pre-swizzle: f32 [K x N] row-major -> packed bf16 WMMA B fragments
// ---------------------------------------------------------------------------
__global__ void pack_b_kernel(const float* __restrict__ src,
                              __bf16* __restrict__ dst, int K, int N) {
  int p = blockIdx.x * 256 + threadIdx.x;
  if (p < K * N) {
    int e     = p & 15;
    int lane  = (p >> 4) & 31;
    int tile  = p >> 9;
    int numNT = N >> 4;
    int nt    = tile % numNT;
    int kt    = tile / numNT;
    int k     = kt * 32 + (e < 8 ? (lane >> 4) * 8 + e : 16 + (lane >> 4) * 8 + (e - 8));
    int n     = nt * 16 + (lane & 15);
    dst[p]    = (__bf16)src[(size_t)k * N + n];
  }
}

__global__ void embed_s_kernel(const int* __restrict__ S,
                               const float* __restrict__ Wemb,
                               __bf16* __restrict__ out) {
  int i = blockIdx.x * 256 + threadIdx.x;  // over BN*32 4-wide chunks
  if (i < BN * 32) {
    int node = i >> 5;
    int c4   = (i & 31) * 4;
    const float* src = Wemb + (size_t)S[node] * Hh + c4;
    v4bf y;
#pragma unroll
    for (int j = 0; j < 4; ++j) y[j] = (__bf16)src[j];
    *(v4bf*)(out + (size_t)node * Hh + c4) = y;
  }
}

// ---------------------------------------------------------------------------
// proj128: C[M x 128] = A[M x 128](f32) @ Bw(packed bf16 128x128) + bias
// 64 rows per workgroup, 8 waves; wave w owns N-tile w; B reused over 4 M-tiles.
// ---------------------------------------------------------------------------
__global__ void proj128_kernel(const float* __restrict__ A,
                               const __bf16* __restrict__ Bw,
                               const float* __restrict__ bias,
                               float* __restrict__ outF,
                               __bf16* __restrict__ outB) {
  __shared__ __bf16 Al[64 * 152];
  const int tid  = threadIdx.x;
  const int wave = tid >> 5;
  const int lane = tid & 31;
  const size_t rowBase = (size_t)blockIdx.x * 64;

  for (int e = tid; e < 64 * 32; e += 256) {
    int r = e >> 5, c = (e & 31) * 4;
    v4f x = *(const v4f*)(A + (rowBase + r) * 128 + c);
    v4bf y;
#pragma unroll
    for (int j = 0; j < 4; ++j) y[j] = (__bf16)x[j];
    *(v4bf*)(Al + r * 152 + c) = y;
  }
  __syncthreads();

  const int nb = wave;
  v8f acc[4] = {};
#pragma unroll
  for (int kk = 0; kk < 4; ++kk) {
    v16bf b = load_bp(Bw, 8, kk, nb, lane);
#pragma unroll
    for (int mt = 0; mt < 4; ++mt) {
      v16bf a = load_a(Al, 152, mt * 16, kk * 32, lane);
      acc[mt] = wmma_bf16(a, b, acc[mt]);
    }
  }
  const int half = lane >> 4;
  const int n    = nb * 16 + (lane & 15);
  const float bn = bias[n];
#pragma unroll
  for (int mt = 0; mt < 4; ++mt) {
#pragma unroll
    for (int i = 0; i < 8; ++i) {
      size_t row = rowBase + mt * 16 + half * 8 + i;
      float  x   = acc[mt][i] + bn;
      if (outF) outF[row * 128 + n] = x;
      outB[row * 128 + n] = (__bf16)x;
    }
  }
}

// ---------------------------------------------------------------------------
// Fused message kernel: one workgroup per node. Builds 32 x INW concat matrix
// (rows = K neighbors, padded), 3-layer WMMA MLP, masked reduce over K -> dh.
//   enc (INW=384): [hV(n) | hE(n,k) | hV(idx)]
//   dec (INW=512): [hV(n) | hE(n,k) | ar?hS(idx):0 | ar?hV(idx):hVenc(idx)]
// ---------------------------------------------------------------------------
template <int INW, bool DEC>
__global__ void msg_kernel(const __bf16* __restrict__ hVb,
                           const __bf16* __restrict__ hEb,
                           const __bf16* __restrict__ hSb,
                           const __bf16* __restrict__ hVencb,
                           const int* __restrict__ Eidx,
                           const float* __restrict__ mask,
                           const __bf16* __restrict__ W1, const float* __restrict__ b1,
                           const __bf16* __restrict__ W2, const float* __restrict__ b2,
                           const __bf16* __restrict__ W3, const float* __restrict__ b3,
                           float* __restrict__ dh) {
  constexpr int LDA0 = INW + 24;
  __shared__ __bf16 A0[32 * LDA0];
  __shared__ __bf16 A1[32 * 152];
  __shared__ __bf16 A2[32 * 152];
  __shared__ int    idxs[32];
  __shared__ float  rowm[32];

  const int tid = threadIdx.x;
  const int bn  = blockIdx.x;
  const int b   = bn >> 10;
  const int n   = bn & (Nn - 1);

  if (tid < 32) {
    int   id = (tid < Kk) ? Eidx[(size_t)bn * Kk + tid] : 0;
    float rm;
    if (tid >= Kk) rm = 0.0f;
    else if (DEC)  rm = 1.0f;
    else           rm = mask[b * Nn + n] * mask[b * Nn + id];
    idxs[tid] = id;
    rowm[tid] = rm;
  }
  __syncthreads();

  // Build A0 in 16-byte chunks (segments are 128-col aligned -> chunk-uniform).
  constexpr int CH = INW / 8;
  for (int e = tid; e < 32 * CH; e += 256) {
    int r = e / CH;
    int c = (e - r * CH) * 8;
    v8bf v = {};
    if (r < Kk) {
      int id = idxs[r];
      const __bf16* src = nullptr;
      if (c < 128)       src = hVb + (size_t)bn * 128 + c;
      else if (c < 256)  src = hEb + ((size_t)bn * Kk + r) * 128 + (c - 128);
      else if (!DEC)     src = hVb + ((size_t)b * Nn + id) * 128 + (c - 256);
      else if (c < 384)  { if (id < n) src = hSb + ((size_t)b * Nn + id) * 128 + (c - 256); }
      else               src = (id < n ? hVb : hVencb) + ((size_t)b * Nn + id) * 128 + (c - 384);
      if (src) v = *(const v8bf*)src;
    }
    *(v8bf*)(A0 + (size_t)r * LDA0 + c) = v;
  }
  __syncthreads();

  const int wave = tid >> 5;
  const int lane = tid & 31;
  const int nb   = wave;

  // GEMM1: (32 x INW) @ W1 + b1, ReLU -> A1 (B fragment shared by both M-tiles)
  {
    v8f acc0 = {}, acc1 = {};
#pragma unroll
    for (int kk = 0; kk < INW / 32; ++kk) {
      v16bf bb = load_bp(W1, 8, kk, nb, lane);
      v16bf a0 = load_a(A0, LDA0, 0, kk * 32, lane);
      v16bf a1 = load_a(A0, LDA0, 16, kk * 32, lane);
      acc0 = wmma_bf16(a0, bb, acc0);
      acc1 = wmma_bf16(a1, bb, acc1);
    }
    store_tile_relu(A1, 152, 0, nb * 16, acc0, b1, lane);
    store_tile_relu(A1, 152, 16, nb * 16, acc1, b1, lane);
  }
  __syncthreads();

  // GEMM2: (32 x 128) @ W2 + b2, ReLU -> A2
  {
    v8f acc0 = {}, acc1 = {};
#pragma unroll
    for (int kk = 0; kk < 4; ++kk) {
      v16bf bb = load_bp(W2, 8, kk, nb, lane);
      v16bf a0 = load_a(A1, 152, 0, kk * 32, lane);
      v16bf a1 = load_a(A1, 152, 16, kk * 32, lane);
      acc0 = wmma_bf16(a0, bb, acc0);
      acc1 = wmma_bf16(a1, bb, acc1);
    }
    store_tile_relu(A2, 152, 0, nb * 16, acc0, b2, lane);
    store_tile_relu(A2, 152, 16, nb * 16, acc1, b2, lane);
  }
  __syncthreads();

  // GEMM3: (32 x 128) @ W3 + b3, row-mask, reduce over K -> dh[bn]
  {
    v8f acc0 = {}, acc1 = {};
#pragma unroll
    for (int kk = 0; kk < 4; ++kk) {
      v16bf bb = load_bp(W3, 8, kk, nb, lane);
      v16bf a0 = load_a(A2, 152, 0, kk * 32, lane);
      v16bf a1 = load_a(A2, 152, 16, kk * 32, lane);
      acc0 = wmma_bf16(a0, bb, acc0);
      acc1 = wmma_bf16(a1, bb, acc1);
    }
    const int half  = lane >> 4;
    const int nn    = nb * 16 + (lane & 15);
    const float bn3 = b3[nn];
    float colsum = 0.0f;
#pragma unroll
    for (int i = 0; i < 8; ++i) {
      colsum += (acc0[i] + bn3) * rowm[half * 8 + i];
      colsum += (acc1[i] + bn3) * rowm[16 + half * 8 + i];
    }
    colsum += __shfl_xor(colsum, 16, 32);
    if (lane < 16) dh[(size_t)bn * 128 + nn] = colsum;
  }
}

// ---------------------------------------------------------------------------
// Node update: h = LN1(hV + dh/SCALE); h = LN2(h + FFN(h)); h *= mask
// 16 nodes per workgroup; FFN 128 -> 512 (ReLU) -> 128 via WMMA.
// ---------------------------------------------------------------------------
__global__ void node_update_kernel(float* __restrict__ hVf,
                                   __bf16* __restrict__ hVb,
                                   const float* __restrict__ dh,
                                   const float* __restrict__ mask,
                                   const float* __restrict__ g1, const float* __restrict__ bt1,
                                   const __bf16* __restrict__ Win, const float* __restrict__ bin,
                                   const __bf16* __restrict__ Wff, const float* __restrict__ bff,
                                   const float* __restrict__ g2, const float* __restrict__ bt2) {
  __shared__ float  Hf[16 * 128];
  __shared__ __bf16 Hb[16 * 152];
  __shared__ __bf16 Fm[16 * 536];
  __shared__ float  Fo[16 * 128];

  const int tid = threadIdx.x;
  const int r   = tid >> 4;
  const int cg  = tid & 15;
  const size_t node = (size_t)blockIdx.x * 16 + r;

  // ---- residual + LN1 ----
  float x[8];
  float s = 0.0f, ss = 0.0f;
#pragma unroll
  for (int j = 0; j < 8; ++j) {
    int c = cg * 8 + j;
    float v = hVf[node * 128 + c] + dh[node * 128 + c] * (1.0f / SCALE);
    x[j] = v;
    s += v;
    ss += v * v;
  }
#pragma unroll
  for (int off = 1; off < 16; off <<= 1) {
    s  += __shfl_xor(s, off, 32);
    ss += __shfl_xor(ss, off, 32);
  }
  {
    float mu  = s * (1.0f / 128.0f);
    float var = ss * (1.0f / 128.0f) - mu * mu;
    float inv = rsqrtf(var + EPSC);
#pragma unroll
    for (int j = 0; j < 8; ++j) {
      int c = cg * 8 + j;
      float y = g1[c] * (x[j] - mu) * inv + bt1[c];
      Hf[r * 128 + c] = y;
      Hb[r * 152 + c] = (__bf16)y;
    }
  }
  __syncthreads();

  const int wave = tid >> 5;
  const int lane = tid & 31;

  // ---- FFN GEMM1: (16x128) @ Win + bin, ReLU -> Fm (A shared by 4 N-tiles) ----
  {
    v8f acc[4] = {};
#pragma unroll
    for (int kk = 0; kk < 4; ++kk) {
      v16bf a = load_a(Hb, 152, 0, kk * 32, lane);
#pragma unroll
      for (int t = 0; t < 4; ++t) {
        v16bf b = load_bp(Win, 32, kk, wave + t * 8, lane);
        acc[t]  = wmma_bf16(a, b, acc[t]);
      }
    }
#pragma unroll
    for (int t = 0; t < 4; ++t)
      store_tile_relu(Fm, 536, 0, (wave + t * 8) * 16, acc[t], bin, lane);
  }
  __syncthreads();

  // ---- FFN GEMM2: (16x512) @ Wff + bff -> Fo (f32) ----
  {
    int nb  = wave;
    v8f acc = {};
#pragma unroll
    for (int kk = 0; kk < 16; ++kk) {
      v16bf a = load_a(Fm, 536, 0, kk * 32, lane);
      v16bf b = load_bp(Wff, 8, kk, nb, lane);
      acc = wmma_bf16(a, b, acc);
    }
    const int half = lane >> 4;
    const int n    = nb * 16 + (lane & 15);
    const float bn = bff[n];
#pragma unroll
    for (int i = 0; i < 8; ++i) Fo[(half * 8 + i) * 128 + n] = acc[i] + bn;
  }
  __syncthreads();

  // ---- residual + LN2 + mask ----
  float x2[8];
  s = 0.0f; ss = 0.0f;
#pragma unroll
  for (int j = 0; j < 8; ++j) {
    int c = cg * 8 + j;
    float v = Hf[r * 128 + c] + Fo[r * 128 + c];
    x2[j] = v;
    s += v;
    ss += v * v;
  }
#pragma unroll
  for (int off = 1; off < 16; off <<= 1) {
    s  += __shfl_xor(s, off, 32);
    ss += __shfl_xor(ss, off, 32);
  }
  {
    float mu  = s * (1.0f / 128.0f);
    float var = ss * (1.0f / 128.0f) - mu * mu;
    float inv = rsqrtf(var + EPSC);
    float mk  = mask[node];
#pragma unroll
    for (int j = 0; j < 8; ++j) {
      int c = cg * 8 + j;
      float y = (g2[c] * (x2[j] - mu) * inv + bt2[c]) * mk;
      hVf[node * 128 + c] = y;
      hVb[node * 128 + c] = (__bf16)y;
    }
  }
}

// ---------------------------------------------------------------------------
// Output head: logits = hV @ Wout + bout ; log_softmax over VOCAB=21
// ---------------------------------------------------------------------------
__global__ void out_kernel(const float* __restrict__ hVf,
                           const float* __restrict__ Wout,
                           const float* __restrict__ bout,
                           float* __restrict__ out) {
  const size_t node = (size_t)blockIdx.x * 8 + (threadIdx.x >> 5);
  const int lane = threadIdx.x & 31;

  float logit = -3.4e38f;
  if (lane < VOC) {
    float sum = bout[lane];
    const v4f* row = (const v4f*)(hVf + node * 128);
#pragma unroll 4
    for (int c4 = 0; c4 < 32; ++c4) {
      v4f x = row[c4];
#pragma unroll
      for (int j = 0; j < 4; ++j) sum += x[j] * Wout[(c4 * 4 + j) * VOC + lane];
    }
    logit = sum;
  }
  float m = logit;
#pragma unroll
  for (int off = 16; off >= 1; off >>= 1) m = fmaxf(m, __shfl_xor(m, off, 32));
  float e = (lane < VOC) ? expf(logit - m) : 0.0f;
  float se = e;
#pragma unroll
  for (int off = 16; off >= 1; off >>= 1) se += __shfl_xor(se, off, 32);
  if (lane < VOC) out[node * VOC + lane] = logit - m - logf(se);
}

// ---------------------------------------------------------------------------
// Host orchestration
// ---------------------------------------------------------------------------
extern "C" void kernel_launch(void* const* d_in, const int* in_sizes, int n_in,
                              void* d_out, int out_size, void* d_ws, size_t ws_size,
                              hipStream_t stream) {
  (void)in_sizes; (void)n_in; (void)out_size; (void)ws_size;

  const int*   S     = (const int*)d_in[0];
  const float* V     = (const float*)d_in[1];
  const float* E     = (const float*)d_in[2];
  const int*   Eidx  = (const int*)d_in[3];
  const float* mask  = (const float*)d_in[4];
  const float* Wv    = (const float*)d_in[5];
  const float* bv    = (const float*)d_in[6];
  const float* We    = (const float*)d_in[7];
  const float* be    = (const float*)d_in[8];
  const float* Wemb  = (const float*)d_in[9];
  const float* Wout  = (const float*)d_in[10];
  const float* bout  = (const float*)d_in[11];
  const float* encW1  = (const float*)d_in[12];
  const float* encb1  = (const float*)d_in[13];
  const float* encW2  = (const float*)d_in[14];
  const float* encb2  = (const float*)d_in[15];
  const float* encW3  = (const float*)d_in[16];
  const float* encb3  = (const float*)d_in[17];
  const float* encWin = (const float*)d_in[18];
  const float* encbin = (const float*)d_in[19];
  const float* encWff = (const float*)d_in[20];
  const float* encbff = (const float*)d_in[21];
  const float* encg1  = (const float*)d_in[22];
  const float* encbt1 = (const float*)d_in[23];
  const float* encg2  = (const float*)d_in[24];
  const float* encbt2 = (const float*)d_in[25];
  const float* decW1  = (const float*)d_in[26];
  const float* decb1  = (const float*)d_in[27];
  const float* decW2  = (const float*)d_in[28];
  const float* decb2  = (const float*)d_in[29];
  const float* decW3  = (const float*)d_in[30];
  const float* decb3  = (const float*)d_in[31];
  const float* decWin = (const float*)d_in[32];
  const float* decbin = (const float*)d_in[33];
  const float* decWff = (const float*)d_in[34];
  const float* decbff = (const float*)d_in[35];
  const float* decg1  = (const float*)d_in[36];
  const float* decbt1 = (const float*)d_in[37];
  const float* decg2  = (const float*)d_in[38];
  const float* decbt2 = (const float*)d_in[39];

  // ---- workspace carve ----
  char* wsp = (char*)d_ws;
  size_t off = 0;
  auto carve = [&](size_t bytes) -> char* {
    char* p = wsp + off;
    off += (bytes + 255) & ~(size_t)255;
    return p;
  };
  float*  hVf    = (float*)carve((size_t)BN * Hh * 4);
  float*  dh     = (float*)carve((size_t)BN * Hh * 4);
  __bf16* hVb    = (__bf16*)carve((size_t)BN * Hh * 2);
  __bf16* hVencb = (__bf16*)carve((size_t)BN * Hh * 2);
  __bf16* hSb    = (__bf16*)carve((size_t)BN * Hh * 2);
  __bf16* hEb    = (__bf16*)carve((size_t)BN * Kk * Hh * 2);
  __bf16* Wvb    = (__bf16*)carve((size_t)Hh * Hh * 2);
  __bf16* Web    = (__bf16*)carve((size_t)Hh * Hh * 2);
  __bf16* eW1b   = (__bf16*)carve((size_t)Ll * 384 * Hh * 2);
  __bf16* eW2b   = (__bf16*)carve((size_t)Ll * Hh * Hh * 2);
  __bf16* eW3b   = (__bf16*)carve((size_t)Ll * Hh * Hh * 2);
  __bf16* eWinb  = (__bf16*)carve((size_t)Ll * Hh * 512 * 2);
  __bf16* eWffb  = (__bf16*)carve((size_t)Ll * 512 * Hh * 2);
  __bf16* dW1b   = (__bf16*)carve((size_t)Ll * 512 * Hh * 2);
  __bf16* dW2b   = (__bf16*)carve((size_t)Ll * Hh * Hh * 2);
  __bf16* dW3b   = (__bf16*)carve((size_t)Ll * Hh * Hh * 2);
  __bf16* dWinb  = (__bf16*)carve((size_t)Ll * Hh * 512 * 2);
  __bf16* dWffb  = (__bf16*)carve((size_t)Ll * 512 * Hh * 2);

  auto pack = [&](const float* src, __bf16* dst, int K, int N) {
    int total = K * N;
    pack_b_kernel<<<(total + 255) / 256, 256, 0, stream>>>(src, dst, K, N);
  };
  pack(Wv, Wvb, Hh, Hh);
  pack(We, Web, Hh, Hh);
  for (int l = 0; l < Ll; ++l) {
    pack(encW1 + (size_t)l * 384 * Hh, eW1b + (size_t)l * 384 * Hh, 384, Hh);
    pack(encW2 + (size_t)l * Hh * Hh, eW2b + (size_t)l * Hh * Hh, Hh, Hh);
    pack(encW3 + (size_t)l * Hh * Hh, eW3b + (size_t)l * Hh * Hh, Hh, Hh);
    pack(encWin + (size_t)l * Hh * 512, eWinb + (size_t)l * Hh * 512, Hh, 512);
    pack(encWff + (size_t)l * 512 * Hh, eWffb + (size_t)l * 512 * Hh, 512, Hh);
    pack(decW1 + (size_t)l * 512 * Hh, dW1b + (size_t)l * 512 * Hh, 512, Hh);
    pack(decW2 + (size_t)l * Hh * Hh, dW2b + (size_t)l * Hh * Hh, Hh, Hh);
    pack(decW3 + (size_t)l * Hh * Hh, dW3b + (size_t)l * Hh * Hh, Hh, Hh);
    pack(decWin + (size_t)l * Hh * 512, dWinb + (size_t)l * Hh * 512, Hh, 512);
    pack(decWff + (size_t)l * 512 * Hh, dWffb + (size_t)l * 512 * Hh, 512, Hh);
  }

  embed_s_kernel<<<(BN * 32) / 256, 256, 0, stream>>>(S, Wemb, hSb);

  // h_V = V @ Wv + bv ; h_E = E @ We + be
  proj128_kernel<<<BN / 64, 256, 0, stream>>>(V, Wvb, bv, hVf, hVb);
  proj128_kernel<<<(BN * Kk) / 64, 256, 0, stream>>>(E, Web, be, nullptr, hEb);

  // ---- encoder ----
  for (int l = 0; l < Ll; ++l) {
    msg_kernel<384, false><<<BN, 256, 0, stream>>>(
        hVb, hEb, nullptr, nullptr, Eidx, mask,
        eW1b + (size_t)l * 384 * Hh, encb1 + l * Hh,
        eW2b + (size_t)l * Hh * Hh, encb2 + l * Hh,
        eW3b + (size_t)l * Hh * Hh, encb3 + l * Hh, dh);
    node_update_kernel<<<BN / 16, 256, 0, stream>>>(
        hVf, hVb, dh, mask, encg1 + l * Hh, encbt1 + l * Hh,
        eWinb + (size_t)l * Hh * 512, encbin + l * 512,
        eWffb + (size_t)l * 512 * Hh, encbff + l * Hh,
        encg2 + l * Hh, encbt2 + l * Hh);
  }

  hipMemcpyAsync(hVencb, hVb, (size_t)BN * Hh * 2, hipMemcpyDeviceToDevice, stream);

  // ---- decoder ----
  for (int l = 0; l < Ll; ++l) {
    msg_kernel<512, true><<<BN, 256, 0, stream>>>(
        hVb, hEb, hSb, hVencb, Eidx, mask,
        dW1b + (size_t)l * 512 * Hh, decb1 + l * Hh,
        dW2b + (size_t)l * Hh * Hh, decb2 + l * Hh,
        dW3b + (size_t)l * Hh * Hh, decb3 + l * Hh, dh);
    node_update_kernel<<<BN / 16, 256, 0, stream>>>(
        hVf, hVb, dh, mask, decg1 + l * Hh, decbt1 + l * Hh,
        dWinb + (size_t)l * Hh * 512, decbin + l * 512,
        dWffb + (size_t)l * 512 * Hh, decbff + l * Hh,
        decg2 + l * Hh, decbt2 + l * Hh);
  }

  out_kernel<<<BN / 8, 256, 0, stream>>>(hVf, Wout, bout, (float*)d_out);
}
